// MultiHeadAttention_39676907883348
// MI455X (gfx1250) — matrix operand order repaired and checked by reading, not compiled, not run on hardware
//
#include <hip/hip_runtime.h>

// ---------------------------------------------------------------------------
// MHA for B=8, S=256, D=4096, H=16, HD=256 on gfx1250 (wave32, WMMA).
// bf16 hi/lo split (3-product emulated fp32) GEMMs via v_wmma_f32_16x16x32_bf16.
// Round 2: 64x64 wave tile (4x4), block tile 128x256 -> 0.67 b128 loads per wmma.
// ---------------------------------------------------------------------------

typedef __attribute__((ext_vector_type(16))) __bf16 bf16x16;
typedef __attribute__((ext_vector_type(8)))  float  floatx8;

static constexpr int BB = 8;     // batch
static constexpr int SS = 256;   // seq
static constexpr int DD = 4096;  // model dim
static constexpr int HH = 16;    // heads
static constexpr int HD = 256;   // head dim

#define MODE_QK     0  // proj out -> [B,H,S,HD] hi/lo, +bias, channel mask
#define MODE_V      1  // proj out -> [B,H,HD,S] hi/lo (transposed), +bias
#define MODE_SCORES 2  // batched, fp32 out * scale
#define MODE_CTX    3  // batched, hi/lo out * scale -> [B,S,D]
#define MODE_OUT    4  // fp32 out + bias -> d_out

__device__ __forceinline__ unsigned short f2bf(float f) {
  unsigned u = __builtin_bit_cast(unsigned, f);
  u += 0x7fffu + ((u >> 16) & 1u);   // round-to-nearest-even
  return (unsigned short)(u >> 16);
}
__device__ __forceinline__ float bf2f(unsigned short h) {
  unsigned u = ((unsigned)h) << 16;
  return __builtin_bit_cast(float, u);
}
__device__ __forceinline__ void splitbf(float x, unsigned short& hi, unsigned short& lo) {
  hi = f2bf(x);
  lo = f2bf(x - bf2f(hi));
}

// ---------------------------------------------------------------------------
// fp32 -> bf16 hi/lo split (elementwise)
// ---------------------------------------------------------------------------
__global__ __launch_bounds__(256) void split_kernel(const float* __restrict__ x,
                                                    unsigned short* __restrict__ hi,
                                                    unsigned short* __restrict__ lo,
                                                    long long n) {
  long long i = (long long)blockIdx.x * blockDim.x + threadIdx.x;
  long long stride = (long long)gridDim.x * blockDim.x;
  for (; i < n; i += stride) {
    unsigned short h, l;
    splitbf(x[i], h, l);
    hi[i] = h;
    lo[i] = l;
  }
}

// ---------------------------------------------------------------------------
// Generic hi/lo bf16 GEMM:  C[m,n] = sum_k A[m,k] * B[n,k]   (both K-major)
// Block: 256 threads = 8 waves arranged 2(M) x 4(N); wave tile 64x64 (4x4
// sub-tiles of 16x16); block tile 128x256; K step 32.
// All problem dims divide the tile sizes exactly -> no bounds checks.
// ---------------------------------------------------------------------------
__global__ __launch_bounds__(256) void gemm_bf16x2(
    const unsigned short* __restrict__ AhiP, const unsigned short* __restrict__ AloP,
    long long aStrideZ,
    const unsigned short* __restrict__ BhiP, const unsigned short* __restrict__ BloP,
    long long bStrideZ,
    const float* __restrict__ bias, const int* __restrict__ maskp,
    float* __restrict__ outF,
    unsigned short* __restrict__ outHi, unsigned short* __restrict__ outLo,
    int K, int lda, int ldb, float scale, int mode) {
  const int tid  = threadIdx.x;
  const int lane = tid & 31;
  const int ll   = lane & 15;   // row (A) / col (B,C)
  const int lh   = lane >> 4;   // half select
  const int w    = tid >> 5;    // wave 0..7
  const int wm   = w & 1;       // 0..1  (M direction)
  const int wn   = w >> 1;      // 0..3  (N direction)

  const int z = blockIdx.z;
  const unsigned short* Ah = AhiP + (size_t)z * (size_t)aStrideZ;
  const unsigned short* Al = AloP + (size_t)z * (size_t)aStrideZ;
  const unsigned short* Bh = BhiP + (size_t)z * (size_t)bStrideZ;
  const unsigned short* Bl = BloP + (size_t)z * (size_t)bStrideZ;

  const int mBase = blockIdx.y * 128 + wm * 64;  // 4 M-tiles of 16
  const int nBase = blockIdx.x * 256 + wn * 64;  // 4 N-tiles of 16

  floatx8 acc[4][4] = {};

  for (int kb = 0; kb < K; kb += 32) {
    bf16x16 aH[4], aL[4], bH[4], bL[4];

    // A fragment (16x32 bf16): lane ll = row; k chunks [lh*8, +8) and [16+lh*8, +8)
#pragma unroll
    for (int mt = 0; mt < 4; ++mt) {
      size_t ro = (size_t)(mBase + mt * 16 + ll) * (size_t)lda + (size_t)kb + (size_t)(lh * 8);
      uint4* d = (uint4*)&aH[mt];
      d[0] = *(const uint4*)(Ah + ro);
      d[1] = *(const uint4*)(Ah + ro + 16);
      uint4* e = (uint4*)&aL[mt];
      e[0] = *(const uint4*)(Al + ro);
      e[1] = *(const uint4*)(Al + ro + 16);
    }
    // B fragment (32x16 bf16, N x K rows): lane ll = col; k chunks [lh*8, +8) and [16+lh*8, +8)
#pragma unroll
    for (int nt = 0; nt < 4; ++nt) {
      size_t co = (size_t)(nBase + nt * 16 + ll) * (size_t)ldb + (size_t)kb + (size_t)(lh * 8);
      uint4* d = (uint4*)&bH[nt];
      d[0] = *(const uint4*)(Bh + co);
      d[1] = *(const uint4*)(Bh + co + 16);
      uint4* e = (uint4*)&bL[nt];
      e[0] = *(const uint4*)(Bl + co);
      e[1] = *(const uint4*)(Bl + co + 16);
    }

    // 3-product hi/lo emulation: hi*hi + hi*lo + lo*hi (lo*lo dropped)
#pragma unroll
    for (int mt = 0; mt < 4; ++mt) {
#pragma unroll
      for (int nt = 0; nt < 4; ++nt) {
        acc[mt][nt] = __builtin_amdgcn_wmma_f32_16x16x32_bf16(
            false, aH[mt], false, bH[nt], (short)0, acc[mt][nt], false, false);
        acc[mt][nt] = __builtin_amdgcn_wmma_f32_16x16x32_bf16(
            false, aH[mt], false, bL[nt], (short)0, acc[mt][nt], false, false);
        acc[mt][nt] = __builtin_amdgcn_wmma_f32_16x16x32_bf16(
            false, aL[mt], false, bH[nt], (short)0, acc[mt][nt], false, false);
      }
    }
  }

  // Epilogue. C layout: element r of v8f -> (M = tile + r + 8*lh, N = tile + ll)
#pragma unroll
  for (int mt = 0; mt < 4; ++mt) {
#pragma unroll
    for (int nt = 0; nt < 4; ++nt) {
#pragma unroll
      for (int r = 0; r < 8; ++r) {
        int row = mBase + mt * 16 + lh * 8 + r;
        int col = nBase + nt * 16 + ll;
        float val = acc[mt][nt][r];
        if (mode == MODE_QK) {
          val += bias[col];
          int b = row >> 8, s = row & 255;
          int h = col >> 8, hd = col & 255;
          if (maskp[b * 256 + hd] == 0) val = 0.0f;  // reference channel mask
          size_t o = (((size_t)(b * 16 + h) * 256 + (size_t)s) * 256 + (size_t)hd);
          unsigned short hi_, lo_;
          splitbf(val, hi_, lo_);
          outHi[o] = hi_;
          outLo[o] = lo_;
        } else if (mode == MODE_V) {
          val += bias[col];
          int b = row >> 8, s = row & 255;
          int h = col >> 8, hd = col & 255;
          // transposed store [B,H,HD,S] so P@V has K-contiguous B-matrix
          size_t o = (((size_t)(b * 16 + h) * 256 + (size_t)hd) * 256 + (size_t)s);
          unsigned short hi_, lo_;
          splitbf(val, hi_, lo_);
          outHi[o] = hi_;
          outLo[o] = lo_;
        } else if (mode == MODE_SCORES) {
          outF[((size_t)z * 256 + (size_t)row) * 256 + (size_t)col] = val * scale;
        } else if (mode == MODE_CTX) {
          int b = z >> 4, h = z & 15;
          val *= scale;  // D^-0.5 folded here
          size_t o = ((size_t)(b * 256 + row) * 4096) + (size_t)(h * 256 + col);
          unsigned short hi_, lo_;
          splitbf(val, hi_, lo_);
          outHi[o] = hi_;
          outLo[o] = lo_;
        } else {  // MODE_OUT
          outF[(size_t)row * 4096 + (size_t)col] = val + bias[col];
        }
      }
    }
  }
}

// ---------------------------------------------------------------------------
// Row softmax over 256 columns; one wave per row; emits bf16 hi/lo P.
// ---------------------------------------------------------------------------
__global__ __launch_bounds__(256) void softmax_kernel(const float* __restrict__ s,
                                                      unsigned short* __restrict__ phi,
                                                      unsigned short* __restrict__ plo) {
  int lane = threadIdx.x & 31;
  long long warp = ((long long)blockIdx.x * blockDim.x + threadIdx.x) >> 5;
  const float* rowp = s + warp * 256;

  float v[8];
  float mx = -3.0e38f;
#pragma unroll
  for (int i = 0; i < 8; ++i) {
    v[i] = rowp[lane + i * 32];
    mx = fmaxf(mx, v[i]);
  }
#pragma unroll
  for (int off = 16; off > 0; off >>= 1) mx = fmaxf(mx, __shfl_xor(mx, off, 32));

  float sum = 0.0f;
#pragma unroll
  for (int i = 0; i < 8; ++i) {
    v[i] = expf(v[i] - mx);
    sum += v[i];
  }
#pragma unroll
  for (int off = 16; off > 0; off >>= 1) sum += __shfl_xor(sum, off, 32);

  float inv = 1.0f / sum;
#pragma unroll
  for (int i = 0; i < 8; ++i) {
    unsigned short h, l;
    splitbf(v[i] * inv, h, l);
    phi[warp * 256 + lane + i * 32] = h;
    plo[warp * 256 + lane + i * 32] = l;
  }
}

// ---------------------------------------------------------------------------
// Host-side orchestration
// ---------------------------------------------------------------------------
extern "C" void kernel_launch(void* const* d_in, const int* in_sizes, int n_in,
                              void* d_out, int out_size, void* d_ws, size_t ws_size,
                              hipStream_t stream) {
  const float* q   = (const float*)d_in[0];
  const float* k   = (const float*)d_in[1];
  const float* v   = (const float*)d_in[2];
  const int*  mask = (const int*) d_in[3];
  const float* Wq  = (const float*)d_in[4];
  const float* bq  = (const float*)d_in[5];
  const float* Wk  = (const float*)d_in[6];
  const float* bk  = (const float*)d_in[7];
  const float* Wv  = (const float*)d_in[8];
  const float* bv  = (const float*)d_in[9];
  const float* Wo  = (const float*)d_in[10];
  const float* bo  = (const float*)d_in[11];
  float* out = (float*)d_out;

  const size_t nAct = (size_t)BB * SS * DD;       // 8Mi
  const size_t nW   = (size_t)DD * DD;            // 16Mi
  const size_t nSc  = (size_t)BB * HH * SS * SS;  // 8Mi

  char* ws = (char*)d_ws;
  size_t off = 0;
  auto take = [&](size_t bytes) -> char* {
    char* p = ws + off;
    off += (bytes + 255) & ~(size_t)255;
    return p;
  };

  unsigned short* qhi = (unsigned short*)take(nAct * 2);
  unsigned short* qlo = (unsigned short*)take(nAct * 2);
  unsigned short* khi = (unsigned short*)take(nAct * 2);
  unsigned short* klo = (unsigned short*)take(nAct * 2);
  unsigned short* vhi = (unsigned short*)take(nAct * 2);
  unsigned short* vlo = (unsigned short*)take(nAct * 2);
  unsigned short* wqhi = (unsigned short*)take(nW * 2);
  unsigned short* wqlo = (unsigned short*)take(nW * 2);
  unsigned short* wkhi = (unsigned short*)take(nW * 2);
  unsigned short* wklo = (unsigned short*)take(nW * 2);
  unsigned short* wvhi = (unsigned short*)take(nW * 2);
  unsigned short* wvlo = (unsigned short*)take(nW * 2);
  unsigned short* wohi = (unsigned short*)take(nW * 2);
  unsigned short* wolo = (unsigned short*)take(nW * 2);
  unsigned short* Qhi = (unsigned short*)take(nAct * 2);  // [B,H,S,HD]
  unsigned short* Qlo = (unsigned short*)take(nAct * 2);
  unsigned short* Khi = (unsigned short*)take(nAct * 2);  // [B,H,S,HD]
  unsigned short* Klo = (unsigned short*)take(nAct * 2);
  unsigned short* Vhi = (unsigned short*)take(nAct * 2);  // [B,H,HD,S]
  unsigned short* Vlo = (unsigned short*)take(nAct * 2);
  float*          scF = (float*)take(nSc * 4);            // [B,H,S,S]
  unsigned short* Phi = (unsigned short*)take(nSc * 2);
  unsigned short* Plo = (unsigned short*)take(nSc * 2);
  unsigned short* Chi = (unsigned short*)take(nAct * 2);  // ctx [B,S,D]
  unsigned short* Clo = (unsigned short*)take(nAct * 2);

  // 1) fp32 -> bf16 hi/lo splits
  split_kernel<<<4096, 256, 0, stream>>>(q, qhi, qlo, (long long)nAct);
  split_kernel<<<4096, 256, 0, stream>>>(k, khi, klo, (long long)nAct);
  split_kernel<<<4096, 256, 0, stream>>>(v, vhi, vlo, (long long)nAct);
  split_kernel<<<8192, 256, 0, stream>>>(Wq, wqhi, wqlo, (long long)nW);
  split_kernel<<<8192, 256, 0, stream>>>(Wk, wkhi, wklo, (long long)nW);
  split_kernel<<<8192, 256, 0, stream>>>(Wv, wvhi, wvlo, (long long)nW);
  split_kernel<<<8192, 256, 0, stream>>>(Wo, wohi, wolo, (long long)nW);

  // 2) projections (M=2048, N=4096, K=4096), block tile 128x256
  dim3 gProj(DD / 256, (BB * SS) / 128, 1);
  gemm_bf16x2<<<gProj, 256, 0, stream>>>(qhi, qlo, 0, wqhi, wqlo, 0, bq, mask,
                                         nullptr, Qhi, Qlo, DD, DD, DD, 1.0f, MODE_QK);
  gemm_bf16x2<<<gProj, 256, 0, stream>>>(khi, klo, 0, wkhi, wklo, 0, bk, mask,
                                         nullptr, Khi, Klo, DD, DD, DD, 1.0f, MODE_QK);
  gemm_bf16x2<<<gProj, 256, 0, stream>>>(vhi, vlo, 0, wvhi, wvlo, 0, bv, nullptr,
                                         nullptr, Vhi, Vlo, DD, DD, DD, 1.0f, MODE_V);

  // 3) scores = Qm @ Km^T / sqrt(HD), batched over B*H = 128
  dim3 gAttn(SS / 256, SS / 128, BB * HH);
  gemm_bf16x2<<<gAttn, 256, 0, stream>>>(Qhi, Qlo, (long long)SS * HD,
                                         Khi, Klo, (long long)SS * HD,
                                         nullptr, nullptr, scF, nullptr, nullptr,
                                         HD, HD, HD, 0.0625f, MODE_SCORES);

  // 4) softmax rows (B*H*S = 32768 rows, 256 cols)
  softmax_kernel<<<(BB * HH * SS) / 8, 256, 0, stream>>>(scF, Phi, Plo);

  // 5) ctx = P @ V, scaled by D^-0.5, written as [B,S,D]
  gemm_bf16x2<<<gAttn, 256, 0, stream>>>(Phi, Plo, (long long)SS * SS,
                                         Vhi, Vlo, (long long)HD * SS,
                                         nullptr, nullptr, nullptr, Chi, Clo,
                                         SS, SS, SS, 0.015625f, MODE_CTX);

  // 6) out = ctx @ Wo^T + bo  (fp32 to d_out)
  gemm_bf16x2<<<gProj, 256, 0, stream>>>(Chi, Clo, 0, wohi, wolo, 0, bo, nullptr,
                                         out, nullptr, nullptr, DD, DD, DD, 1.0f, MODE_OUT);
}
